// GRULight_38311108280991
// MI455X (gfx1250) — matrix operand order repaired and checked by reading, not compiled, not run on hardware
//
#include <hip/hip_runtime.h>
#include <math.h>

// ---------------------------------------------------------------------------
// D-MPNN (ChemProp-style) on MI455X / gfx1250.
//  - All GEMMs via v_wmma_f32_16x16x32_bf16 (bf16 inputs, fp32 accumulate).
//  - Messages stored bf16 (halves random-gather traffic; workload is
//    bandwidth-bound at 23.3 TB/s once WMMA handles the ~212 GFLOP).
//  - GRU weights DMA-staged into LDS per block with
//    global_load_async_to_lds_b128 (ASYNCcnt), double-buffered so the DMA for
//    gate-tile j+1 overlaps the WMMAs of gate-tile j. 4 waves share each
//    24 KB stage -> ~4x cut of the 19.6 GB weight-fragment L2 traffic.
//  - h tile staged in padded LDS (stride 132 floats), intra-wave visibility
//    via inline `s_wait_dscnt 0`.
// ---------------------------------------------------------------------------

typedef __attribute__((ext_vector_type(16))) __bf16 v16bf;
typedef __attribute__((ext_vector_type(8)))  __bf16 v8bf;
typedef __attribute__((ext_vector_type(4)))  __bf16 v4bf;
typedef __attribute__((ext_vector_type(8)))  float  v8f;

#define DD   128                    // node/edge feature dim
#define HH   256                    // readout hidden
#define LDSS 132                    // padded LDS row stride (floats)
#define WPB  4                      // waves (16-row M tiles) per block

#define WIH_BYTES   98304           // 24*4*32*16 bf16 fragments per matrix
#define WLDS_BUF    24576           // one gate-tile j: 6 srcs * 4 tk * 1024 B
#define HTILE_BYTES (16 * LDSS * 4) // 8448 B per wave
#define GRU_LDS_TOTAL (2 * WLDS_BUF + WPB * HTILE_BYTES)   // 82944 B dynamic

__device__ __forceinline__ v8f wmma_bf16(v16bf a, v16bf b, v8f c) {
    // D = A(16x32 bf16) * B(32x16 bf16) + C(16x16 f32)
    return __builtin_amdgcn_wmma_f32_16x16x32_bf16(
        false, a, false, b, (short)0, c, false, false);
}

__device__ __forceinline__ float sigmoid_(float x) {
    return 1.0f / (1.0f + __expf(-x));
}

// ---------------------------------------------------------------------------
// Pack fp32 weights into bf16 WMMA B-fragment order:
//   dst[((tn*KT + tk)*32 + lane)*16 + i] = B[k][o],
//   o = tn*16 + (lane&15), k = tk*32 + (lane&16 ? 8 : 0) + (i < 8 ? i : 8 + i).
// transpose=0: B = W (W is [K, NOUT] row-major)
// transpose=1: B[k][o] = W[o][k] (W is [NOUT, K] row-major)
// ---------------------------------------------------------------------------
__global__ void pack_b_kernel(const float* __restrict__ W, __bf16* __restrict__ dst,
                              int K, int NOUT, int transpose, int total)
{
    int idx = blockIdx.x * blockDim.x + threadIdx.x;
    if (idx >= total) return;
    int KT   = K >> 5;
    int i    = idx & 15;
    int lane = (idx >> 4) & 31;
    int tk   = (idx >> 9) % KT;
    int tn   = idx / (512 * KT);
    int o = tn * 16 + (lane & 15);
    int k = tk * 32 + ((lane & 16) ? 8 : 0) + (i < 8 ? i : 8 + i);
    float v = transpose ? W[(size_t)o * K + k] : W[(size_t)k * NOUT + o];
    dst[idx] = (__bf16)v;
}

// ---------------------------------------------------------------------------
// n_input = f_edges[E,32] @ W_edge[32,128], row 0 zeroed, stored bf16.
// One wave per 16-edge tile; single K-tile (K=32), 8 N-tiles.
// ---------------------------------------------------------------------------
__global__ __launch_bounds__(128) void edge_init_kernel(
    const float* __restrict__ f_edges,
    const v16bf* __restrict__ WeP,     // [8][1][32] fragments
    __bf16* __restrict__ ninp)
{
    const int lane = threadIdx.x & 31;
    const int wave = threadIdx.x >> 5;
    const int tileBase = (blockIdx.x * WPB + wave) * 16;
    const int row  = lane & 15;
    const int half = lane >> 4;
    const int kb   = half ? 8 : 0;

    const float* fr = f_edges + (size_t)(tileBase + row) * 32;
    v16bf aA;
#pragma unroll
    for (int i = 0; i < 8; ++i) {
        aA[i]     = (__bf16)fr[kb + i];
        aA[i + 8] = (__bf16)fr[kb + 16 + i];
    }
#pragma unroll
    for (int tn = 0; tn < 8; ++tn) {
        v8f acc = {0.f, 0.f, 0.f, 0.f, 0.f, 0.f, 0.f, 0.f};
        v16bf b = WeP[(size_t)(tn * 32 + lane)];
        acc = wmma_bf16(aA, b, acc);
#pragma unroll
        for (int r = 0; r < 8; ++r) {
            int e = tileBase + r + 8 * half;
            int c = tn * 16 + row;
            float v = acc[r];
            if (e == 0) v = 0.0f;                  // padding edge
            ninp[(size_t)e * DD + c] = (__bf16)v;
        }
    }
}

// ---------------------------------------------------------------------------
// a_message[v] = sum_{s<6} msg[node2edge[v][s]]   (bf16 gather, fp32 sum)
// One wave handles one node's 128 columns (4 cols per lane).
// ---------------------------------------------------------------------------
__global__ __launch_bounds__(256) void aggregate_kernel(
    const __bf16* __restrict__ msg,
    const int* __restrict__ node2edge,
    float* __restrict__ amsg, int Ntot)
{
    int t = blockIdx.x * blockDim.x + threadIdx.x;
    int node = t >> 5;
    if (node >= Ntot) return;
    int c = (t & 31) * 4;
    float a0 = 0.f, a1 = 0.f, a2 = 0.f, a3 = 0.f;
    const int* nb = node2edge + (size_t)node * 6;
#pragma unroll
    for (int s = 0; s < 6; ++s) {
        int e = nb[s];
        v4bf m = *(const v4bf*)(msg + (size_t)e * DD + c);
        a0 += (float)m[0]; a1 += (float)m[1];
        a2 += (float)m[2]; a3 += (float)m[3];
    }
    float4 out; out.x = a0; out.y = a1; out.z = a2; out.w = a3;
    *(float4*)(amsg + (size_t)node * DD + c) = out;
}

// ---------------------------------------------------------------------------
// One GRU message-passing step (one wave per 16-edge tile):
//   h  = amsg[edge2node[e]] - msg_in[b2revb[e]]          (fp32, staged in LDS)
//   r  = sigmoid(gi_r + gh_r), z = sigmoid(gi_z + gh_z)  (gi+gh fused in one C)
//   n  = tanh(gi_n + r * gh_n)
//   msg_out = ((1-z)*n + z*h) * mask(e != 0)             (stored bf16)
// Weight B-fragments for gate-tile j are async-DMA'd into LDS (double buffer)
// and shared by the 4 waves of the block.
// LDS layout: [0, 2*WLDS_BUF) weight buffers; then WPB h tiles.
// ---------------------------------------------------------------------------
__global__ __launch_bounds__(128) void gru_step_kernel(
    const float*  __restrict__ amsg,
    const __bf16* __restrict__ msg_in,
    const __bf16* __restrict__ ninp,
    const int*    __restrict__ edge2node,
    const int*    __restrict__ b2revb,
    const __bf16* __restrict__ Wpack,   // WihP fragments; WhhP at +WIH_BYTES
    const float*  __restrict__ b_ih,
    const float*  __restrict__ b_hh,
    __bf16*       __restrict__ msg_out)
{
    extern __shared__ char smem[];

    const int tid  = threadIdx.x;
    const int lane = tid & 31;
    const int wave = tid >> 5;
    const int tileBase = (blockIdx.x * WPB + wave) * 16;
    const int row  = lane & 15;
    const int half = lane >> 4;

    float* hsh = (float*)(smem + 2 * WLDS_BUF + wave * HTILE_BYTES);
    const unsigned long long wBase = (unsigned long long)Wpack;

    // Stage the 24 KB of B-fragments for gate-tile j into LDS buffer `buf`.
    // chunk c (0..23): tk = c&3, s = c>>2; s<3 -> W_ih gate s, else W_hh gate
    // s-3. Each of the 128 threads DMAs 12 x 16B via async load-to-LDS.
    auto stage = [&](int j, int buf) {
        const unsigned ldsBase = (unsigned)(buf * WLDS_BUF);
#pragma unroll
        for (int q = 0; q < 12; ++q) {
            int u   = q * 128 + tid;      // 0..1535 16B units
            int c   = u >> 6;             // 1 KB chunk id
            int off = (u & 63) * 16;      // byte offset in chunk
            int tk  = c & 3;
            int s   = c >> 2;
            int gate = (s < 3) ? s : s - 3;
            unsigned gOff = (s < 3 ? 0u : (unsigned)WIH_BYTES)
                          + (unsigned)(((gate * 8 + j) * 4 + tk) * 1024 + off);
            unsigned lds = ldsBase + (unsigned)(c * 1024 + off);
            asm volatile("global_load_async_to_lds_b128 %0, %1, %2"
                         :: "v"(lds), "v"(gOff), "s"(wBase) : "memory");
        }
    };

    // Kick DMA for gate-tile 0 immediately; it overlaps the h-phase gathers.
    stage(0, 0);

    // ---- Phase 1: h tile -> LDS (lane pair per edge row, 64 cols each) ----
    {
        const int e   = tileBase + row;
        const int src = edge2node[e];
        const int rev = b2revb[e];
        const float*  ap = amsg   + (size_t)src * DD + half * 64;
        const __bf16* mp = msg_in + (size_t)rev * DD + half * 64;
        float* hp = hsh + row * LDSS + half * 64;
#pragma unroll
        for (int c8 = 0; c8 < 8; ++c8) {
            v8bf   m8 = *(const v8bf*)(mp + c8 * 8);
            float4 a0 = *(const float4*)(ap + c8 * 8);
            float4 a1 = *(const float4*)(ap + c8 * 8 + 4);
            float4 h0, h1;
            h0.x = a0.x - (float)m8[0];  h0.y = a0.y - (float)m8[1];
            h0.z = a0.z - (float)m8[2];  h0.w = a0.w - (float)m8[3];
            h1.x = a1.x - (float)m8[4];  h1.y = a1.y - (float)m8[5];
            h1.z = a1.z - (float)m8[6];  h1.w = a1.w - (float)m8[7];
            *(float4*)(hp + c8 * 8)     = h0;
            *(float4*)(hp + c8 * 8 + 4) = h1;
        }
    }
    // intra-wave LDS visibility (lockstep wave32; just drain DS counter)
    asm volatile("s_wait_dscnt 0" ::: "memory");

    // ---- Phase 1b: A fragments (ISA 16-bit A layout) ----
    const int kb = half ? 8 : 0;
    const float*  hr = hsh + row * LDSS;
    const __bf16* nr = ninp + (size_t)(tileBase + row) * DD;
    v16bf hA[4], nA[4];
#pragma unroll
    for (int tk = 0; tk < 4; ++tk) {
        const int k0 = tk * 32 + kb;
        v8bf n0 = *(const v8bf*)(nr + k0);
        v8bf n1 = *(const v8bf*)(nr + k0 + 16);
#pragma unroll
        for (int i = 0; i < 8; ++i) {
            hA[tk][i]     = (__bf16)hr[k0 + i];
            hA[tk][i + 8] = (__bf16)hr[k0 + 16 + i];
            nA[tk][i]     = n0[i];
            nA[tk][i + 8] = n1[i];
        }
    }

    // DMA for j=0 must be complete in ALL waves before anyone reads it.
    asm volatile("s_wait_asynccnt 0" ::: "memory");
    __syncthreads();

    // ---- Phase 2/3: per 16-column gate tile: 24 WMMAs + gate math ----
#pragma unroll 1
    for (int j = 0; j < 8; ++j) {
        const int buf = j & 1;
        if (j < 7) stage(j + 1, buf ^ 1);   // overlap next stage with compute

        const v16bf* wb = (const v16bf*)(smem + buf * WLDS_BUF);

        const int nc = j * 16 + row;               // column within gate block
        const float br  = b_ih[nc]       + b_hh[nc];
        const float bz  = b_ih[128 + nc] + b_hh[128 + nc];
        const float bgi = b_ih[256 + nc];
        const float bgh = b_hh[256 + nc];
        v8f accR  = {br,  br,  br,  br,  br,  br,  br,  br};
        v8f accZ  = {bz,  bz,  bz,  bz,  bz,  bz,  bz,  bz};
        v8f accGi = {bgi, bgi, bgi, bgi, bgi, bgi, bgi, bgi};
        v8f accGh = {bgh, bgh, bgh, bgh, bgh, bgh, bgh, bgh};
#pragma unroll
        for (int tk = 0; tk < 4; ++tk) {
            v16bf bRi = wb[(0 * 4 + tk) * 32 + lane];
            v16bf bZi = wb[(1 * 4 + tk) * 32 + lane];
            v16bf bNi = wb[(2 * 4 + tk) * 32 + lane];
            v16bf bRh = wb[(3 * 4 + tk) * 32 + lane];
            v16bf bZh = wb[(4 * 4 + tk) * 32 + lane];
            v16bf bNh = wb[(5 * 4 + tk) * 32 + lane];
            accR  = wmma_bf16(nA[tk], bRi, accR);   // gi_r
            accR  = wmma_bf16(hA[tk], bRh, accR);   // + gh_r
            accZ  = wmma_bf16(nA[tk], bZi, accZ);   // gi_z
            accZ  = wmma_bf16(hA[tk], bZh, accZ);   // + gh_z
            accGi = wmma_bf16(nA[tk], bNi, accGi);  // gi_n
            accGh = wmma_bf16(hA[tk], bNh, accGh);  // gh_n
        }
#pragma unroll
        for (int r = 0; r < 8; ++r) {
            const int m = r + 8 * half;             // C/D layout row
            const int e = tileBase + m;
            const float hv = hsh[m * LDSS + nc];
            const float rg = sigmoid_(accR[r]);
            const float zg = sigmoid_(accZ[r]);
            const float ng = tanhf(accGi[r] + rg * accGh[r]);
            float o = (1.0f - zg) * ng + zg * hv;
            if (e == 0) o = 0.0f;                   // mask padding edge
            msg_out[(size_t)e * DD + nc] = (__bf16)o;
        }
        // next iteration's buffer ready (own DMAs done + everyone arrived)
        asm volatile("s_wait_asynccnt 0" ::: "memory");
        __syncthreads();
    }
}

// ---------------------------------------------------------------------------
// Readout: out = relu([emb[f_nodes] ; a_message] @ W_ro + b_ro)   [N,256]
// One wave per 16-node tile; K = 256 (4 ktiles from emb, 4 from amsg).
// ---------------------------------------------------------------------------
__global__ __launch_bounds__(128) void readout_kernel(
    const int*   __restrict__ f_nodes,
    const float* __restrict__ emb,
    const float* __restrict__ amsg,
    const v16bf* __restrict__ WroP,    // [16][8][32] fragments
    const float* __restrict__ b_ro,
    float* __restrict__ out, int Ntot)
{
    const int lane = threadIdx.x & 31;
    const int wave = threadIdx.x >> 5;
    const int tileBase = (blockIdx.x * WPB + wave) * 16;
    if (tileBase >= Ntot) return;                  // N/16 not divisible by WPB
    const int row  = lane & 15;
    const int half = lane >> 4;
    const int kb   = half ? 8 : 0;

    const int v = tileBase + row;
    const float* er = emb  + (size_t)f_nodes[v] * DD;
    const float* ar = amsg + (size_t)v * DD;

    v16bf aA[8];
#pragma unroll
    for (int tk = 0; tk < 8; ++tk) {
        const float* sp = (tk < 4) ? (er + tk * 32) : (ar + (tk - 4) * 32);
#pragma unroll
        for (int i = 0; i < 8; ++i) {
            aA[tk][i]     = (__bf16)sp[kb + i];
            aA[tk][i + 8] = (__bf16)sp[kb + 16 + i];
        }
    }
#pragma unroll 1
    for (int tn = 0; tn < 16; ++tn) {
        const float bias = b_ro[tn * 16 + row];
        v8f acc = {bias, bias, bias, bias, bias, bias, bias, bias};
#pragma unroll
        for (int tk = 0; tk < 8; ++tk)
            acc = wmma_bf16(aA[tk], WroP[(size_t)((tn * 8 + tk) * 32 + lane)], acc);
#pragma unroll
        for (int r = 0; r < 8; ++r) {
            int vr = tileBase + r + 8 * half;
            out[(size_t)vr * HH + tn * 16 + row] = fmaxf(acc[r], 0.0f);
        }
    }
}

// ---------------------------------------------------------------------------
extern "C" void kernel_launch(void* const* d_in, const int* in_sizes, int n_in,
                              void* d_out, int out_size, void* d_ws, size_t ws_size,
                              hipStream_t stream)
{
    const int*   f_nodes   = (const int*)  d_in[0];
    const float* f_edges   = (const float*)d_in[1];
    const int*   node2edge = (const int*)  d_in[2];
    const int*   edge2node = (const int*)  d_in[3];
    const int*   b2revb    = (const int*)  d_in[4];
    const float* emb       = (const float*)d_in[5];
    const float* W_edge    = (const float*)d_in[6];
    const float* W_ih      = (const float*)d_in[7];
    const float* W_hh      = (const float*)d_in[8];
    const float* b_ih      = (const float*)d_in[9];
    const float* b_hh      = (const float*)d_in[10];
    const float* W_ro      = (const float*)d_in[11];
    const float* b_ro      = (const float*)d_in[12];
    float* out = (float*)d_out;

    const int N = in_sizes[0];   // 100000 (divisible by 16)
    const int E = in_sizes[3];   // 400000 (divisible by 64)

    // ---- workspace layout (~359 MB) ----
    char* ws = (char*)d_ws;
    const size_t szMsg  = (size_t)E * DD * sizeof(__bf16);   // 102.4 MB each
    const size_t szAmsg = (size_t)N * DD * sizeof(float);    // 51.2 MB
    __bf16* ninp = (__bf16*)(ws);
    __bf16* msgP = (__bf16*)(ws + szMsg);
    __bf16* msgQ = (__bf16*)(ws + 2 * szMsg);
    float*  amsg = (float*) (ws + 3 * szMsg);
    char*   wbase = ws + 3 * szMsg + szAmsg;
    const int nWih = 24 * 4 * 32 * 16;   // 49152 bf16 elems (= WIH_BYTES/2)
    const int nWro = 16 * 8 * 32 * 16;   // 65536
    const int nWe  =  8 * 1 * 32 * 16;   // 4096
    __bf16* WihP = (__bf16*)(wbase);                              // contiguous:
    __bf16* WhhP = (__bf16*)(wbase + (size_t)nWih * 2);           // WihP+WIH_BYTES
    __bf16* WroP = (__bf16*)(wbase + (size_t)nWih * 4);
    __bf16* WeP  = (__bf16*)(wbase + (size_t)nWih * 4 + (size_t)nWro * 2);

    // ---- pack weights to bf16 WMMA fragments ----
    pack_b_kernel<<<(nWih + 255) / 256, 256, 0, stream>>>(W_ih,   WihP, 128, 384, 1, nWih);
    pack_b_kernel<<<(nWih + 255) / 256, 256, 0, stream>>>(W_hh,   WhhP, 128, 384, 1, nWih);
    pack_b_kernel<<<(nWro + 255) / 256, 256, 0, stream>>>(W_ro,   WroP, 256, 256, 0, nWro);
    pack_b_kernel<<<(nWe  + 255) / 256, 256, 0, stream>>>(W_edge, WeP,   32, 128, 0, nWe);

    // ---- n_input (initial message) ----
    edge_init_kernel<<<E / (16 * WPB), 128, 0, stream>>>(f_edges, (const v16bf*)WeP, ninp);

    // ---- DEPTH-1 = 4 message-passing steps (ping-pong msg buffers) ----
    const __bf16* cur = ninp;
    __bf16* bufs[2] = {msgP, msgQ};
    for (int it = 0; it < 4; ++it) {
        aggregate_kernel<<<((size_t)N * 32 + 255) / 256, 256, 0, stream>>>(
            cur, node2edge, amsg, N);
        __bf16* nxt = bufs[it & 1];
        gru_step_kernel<<<E / (16 * WPB), 128, GRU_LDS_TOTAL, stream>>>(
            amsg, cur, ninp, edge2node, b2revb,
            WihP, b_ih, b_hh, nxt);
        cur = nxt;
    }

    // ---- final aggregation + fused readout ----
    aggregate_kernel<<<((size_t)N * 32 + 255) / 256, 256, 0, stream>>>(
        cur, node2edge, amsg, N);
    const int nTiles = N / 16;
    readout_kernel<<<(nTiles + WPB - 1) / WPB, 128, 0, stream>>>(
        f_nodes, emb, amsg, (const v16bf*)WroP, b_ro, out, N);
}
